// AttnHead_57853209477545
// MI455X (gfx1250) — matrix operand order repaired and checked by reading, not compiled, not run on hardware
//
#include <hip/hip_runtime.h>

#define B_ 8
#define N_ 4096
#define D_ 256

typedef __attribute__((ext_vector_type(16))) _Float16 v16h;
typedef __attribute__((ext_vector_type(8)))  _Float16 h8;
typedef __attribute__((ext_vector_type(8)))  float    v8f;
typedef __attribute__((ext_vector_type(4)))  float    f4;

// ---------------------------------------------------------------------------
// Kernel 1: transpose W_fts (D x D, row-major [e][d]) into f16 Wt [d][e]
// so WMMA A-fragments for GEMM1 are two contiguous 16B loads per lane.
// ---------------------------------------------------------------------------
__global__ void prep_w_kernel(const float* __restrict__ W, _Float16* __restrict__ Wt) {
  int idx = blockIdx.x * 256 + threadIdx.x;   // over D_*D_
  int d = idx >> 8, e = idx & 255;
  Wt[d * D_ + e] = (_Float16)W[e * D_ + d];
}

// ---------------------------------------------------------------------------
// Kernel 2: row projections. One wave (32 lanes) per row of seq.
//   aArr  = seq . w_f1 + b_f1           (the "f1" logits, f32)
//   bRaw  = seq . w_f2 + b_f2
//   Bp    = exp(bRaw), Bn = exp(0.01*bRaw)   (column exp factors)
// ---------------------------------------------------------------------------
__global__ void row_proj_kernel(const float* __restrict__ seq,
                                const float* __restrict__ w1, const float* __restrict__ b1,
                                const float* __restrict__ w2, const float* __restrict__ b2,
                                float* __restrict__ aArr, float* __restrict__ bRaw,
                                float* __restrict__ Bp, float* __restrict__ Bn) {
  int row  = (blockIdx.x * 256 + threadIdx.x) >> 5;   // 0 .. B_*N_-1
  int lane = threadIdx.x & 31;
  const float* rp = seq + (size_t)row * D_;
  float s1 = 0.f, s2 = 0.f;
#pragma unroll
  for (int rep = 0; rep < 2; ++rep) {
    int base = lane * 4 + rep * 128;
    f4 x = *(const f4*)(rp + base);
    f4 u = *(const f4*)(w1 + base);
    f4 v = *(const f4*)(w2 + base);
    s1 += x[0]*u[0] + x[1]*u[1] + x[2]*u[2] + x[3]*u[3];
    s2 += x[0]*v[0] + x[1]*v[1] + x[2]*v[2] + x[3]*v[3];
  }
#pragma unroll
  for (int off = 16; off > 0; off >>= 1) {
    s1 += __shfl_down(s1, off, 32);
    s2 += __shfl_down(s2, off, 32);
  }
  if (lane == 0) {
    float a = s1 + b1[0];
    float b = s2 + b2[0];
    aArr[row] = a;
    bRaw[row] = b;
    Bp[row] = __expf(b);
    Bn[row] = __expf(0.01f * b);
  }
}

// ---------------------------------------------------------------------------
// Kernel 3: per-batch stabilization shift c_b = lrelu(max(f1) + max(f2)).
// Any constant >= all logits works; it cancels exactly in softmax.
// ---------------------------------------------------------------------------
__global__ void batch_reduce_kernel(const float* __restrict__ aArr,
                                    const float* __restrict__ bRaw,
                                    float* __restrict__ cB) {
  __shared__ float sa[256], sb[256];
  int b = blockIdx.x, t = threadIdx.x;
  float ma = -1e30f, mb = -1e30f;
  for (int n = t; n < N_; n += 256) {
    ma = fmaxf(ma, aArr[b * N_ + n]);
    mb = fmaxf(mb, bRaw[b * N_ + n]);
  }
  sa[t] = ma; sb[t] = mb;
  __syncthreads();
  for (int s = 128; s > 0; s >>= 1) {
    if (t < s) { sa[t] = fmaxf(sa[t], sa[t + s]); sb[t] = fmaxf(sb[t], sb[t + s]); }
    __syncthreads();
  }
  if (t == 0) {
    float c = sa[0] + sb[0];
    cB[b] = (c > 0.f) ? c : 0.01f * c;
  }
}

// ---------------------------------------------------------------------------
// Kernel 4: GEMM1 via WMMA f16: SFt[b][d][j] = (seq[b] @ W)^T, stored f16
// TRANSPOSED so kernel 5's B-fragments are contiguous along j.
// Out^T = W^T x seq^T: A = Wt tile (16 d-rows x 32 e), B = seq tile (32 e x 16 j).
// One wave per 16x16 output tile; 8 K-steps of v_wmma_f32_16x16x32_f16.
// ---------------------------------------------------------------------------
__global__ void gemm1_kernel(const float* __restrict__ seq,
                             const _Float16* __restrict__ Wt,
                             _Float16* __restrict__ SFt) {
  int wave = (blockIdx.x * 256 + threadIdx.x) >> 5;   // 0 .. B_*4096-1
  int lane = threadIdx.x & 31;
  int hi = lane >> 4, ln = lane & 15;
  int b   = wave >> 12;
  int rem = wave & 4095;
  int d0 = (rem >> 8) * 16;     // 16 d-tiles
  int j0 = (rem & 255) * 16;    // 256 j-tiles
  const float* seqB = seq + (size_t)b * N_ * D_;

  v8f acc = {0,0,0,0,0,0,0,0};
  for (int e0 = 0; e0 < D_; e0 += 32) {
    // A fragment (16-bit A 16x32): lane row = d0+ln; lo lanes K 0..15, hi lanes 16..31
    int kbA = e0 + hi * 16;
    const _Float16* arow = Wt + (size_t)(d0 + ln) * D_ + kbA;
    v16h A;
    *(h8*)&A       = *(const h8*)arow;
    ((h8*)&A)[1]   = *(const h8*)(arow + 8);
    // B fragment (16-bit B 32x16): lane col = j0+ln; lo lanes K 0..15, hi lanes 16..31
    int kbB = e0 + hi * 16;
    const float* brow = seqB + (size_t)(j0 + ln) * D_ + kbB;
    f4 xs[4];
    xs[0] = *(const f4*)(brow);
    xs[1] = *(const f4*)(brow + 4);
    xs[2] = *(const f4*)(brow + 8);
    xs[3] = *(const f4*)(brow + 12);
    v16h Bv;
#pragma unroll
    for (int g = 0; g < 4; ++g)
#pragma unroll
      for (int t = 0; t < 4; ++t)
        Bv[g * 4 + t] = (_Float16)xs[g][t];
    acc = __builtin_amdgcn_wmma_f32_16x16x32_f16(false, A, false, Bv,
                                                 (short)0, acc, false, false);
  }
  // C/D layout: VGPR r -> row r (lanes 0-15) / r+8 (lanes 16-31), col = lane&15
  _Float16* outp = SFt + ((size_t)b * D_ + d0) * N_ + j0 + ln;
#pragma unroll
  for (int r = 0; r < 8; ++r)
    outp[(size_t)(r + 8 * hi) * N_] = (_Float16)acc[r];
}

// ---------------------------------------------------------------------------
// Kernel 5: fused attention. Block = 8 waves = 32 attention rows x D=256.
// Wave (itile, dq): rows rb+itile*16.., cols dq*64..dq*64+63 (4 WMMA tiles).
// Inner loop over j in steps of 32: build P tile (16x32 f16) directly in the
// A-fragment layout from separable exp factors, 4 x v_wmma_f32_16x16x32_f16,
// accumulate softmax denominator Z in VALU. Normalize + bias + relu epilogue.
// ---------------------------------------------------------------------------
__global__ void attn_kernel(const float* __restrict__ aArr, const float* __restrict__ Bp,
                            const float* __restrict__ Bn, const float* __restrict__ cB,
                            const _Float16* __restrict__ SFt, const float* __restrict__ biasP,
                            float* __restrict__ out) {
  int lane = threadIdx.x & 31;
  int wv   = threadIdx.x >> 5;
  int hi = lane >> 4, ln = lane & 15;
  int b  = blockIdx.x >> 7;              // 128 blocks per batch
  int rb = (blockIdx.x & 127) << 5;      // 32 rows per block
  int itile = wv >> 2;                   // 0/1 : which 16-row tile
  int dq    = wv & 3;                    // 0..3 : 64-col quarter of D

  int irow = rb + itile * 16 + ln;       // this lane's attention row (A row = lane&15)
  float a   = aArr[b * N_ + irow];
  float c   = cB[b];
  float Ap  = __expf(a - c);             // row factor, positive branch (shifted)
  float An  = __expf(0.01f * a - c);     // row factor, negative branch
  float ENA = __expf(-a);                // sign test: logit>0  <=>  Bp_j > e^{-a_i}

  const float* BpB = Bp + b * N_;
  const float* BnB = Bn + b * N_;
  const _Float16* SF = SFt + (size_t)b * D_ * N_;
  int dcol = dq * 64 + ln;

  v8f acc0 = {0,0,0,0,0,0,0,0};
  v8f acc1 = acc0, acc2 = acc0, acc3 = acc0;
  float z = 0.f;

  for (int k0 = 0; k0 < N_; k0 += 32) {
    // Column factors at this lane's A-fragment K indices: kb+{0..15}
    int kb = k0 + hi * 16;
    f4 bpv[4], bnv[4];
    bpv[0] = *(const f4*)(BpB + kb);      bnv[0] = *(const f4*)(BnB + kb);
    bpv[1] = *(const f4*)(BpB + kb + 4);  bnv[1] = *(const f4*)(BnB + kb + 4);
    bpv[2] = *(const f4*)(BpB + kb + 8);  bnv[2] = *(const f4*)(BnB + kb + 8);
    bpv[3] = *(const f4*)(BpB + kb + 12); bnv[3] = *(const f4*)(BnB + kb + 12);

    v16h A;
#pragma unroll
    for (int g = 0; g < 4; ++g)
#pragma unroll
      for (int t = 0; t < 4; ++t) {
        float pb  = bpv[g][t];
        bool pos  = pb > ENA;
        float p   = (pos ? Ap : An) * (pos ? pb : bnv[g][t]);
        z += p;                            // softmax denominator partial
        A[g * 4 + t] = (_Float16)p;
      }

    // B fragments from transposed seq_fts: row d fixed per lane, K=j contiguous
    int kbB = k0 + hi * 16;
    const _Float16* r0 = SF + (size_t)dcol * N_ + kbB;
    v16h B0; *(h8*)&B0 = *(const h8*)r0;                  ((h8*)&B0)[1] = *(const h8*)(r0 + 8);
    acc0 = __builtin_amdgcn_wmma_f32_16x16x32_f16(false, A, false, B0, (short)0, acc0, false, false);
    const _Float16* r1 = r0 + (size_t)16 * N_;
    v16h B1; *(h8*)&B1 = *(const h8*)r1;                  ((h8*)&B1)[1] = *(const h8*)(r1 + 8);
    acc1 = __builtin_amdgcn_wmma_f32_16x16x32_f16(false, A, false, B1, (short)0, acc1, false, false);
    const _Float16* r2 = r0 + (size_t)32 * N_;
    v16h B2; *(h8*)&B2 = *(const h8*)r2;                  ((h8*)&B2)[1] = *(const h8*)(r2 + 8);
    acc2 = __builtin_amdgcn_wmma_f32_16x16x32_f16(false, A, false, B2, (short)0, acc2, false, false);
    const _Float16* r3 = r0 + (size_t)48 * N_;
    v16h B3; *(h8*)&B3 = *(const h8*)r3;                  ((h8*)&B3)[1] = *(const h8*)(r3 + 8);
    acc3 = __builtin_amdgcn_wmma_f32_16x16x32_f16(false, A, false, B3, (short)0, acc3, false, false);
  }

  // Lanes l and l^16 each hold half of row (l&15)'s sum -> combine (wave32 xor-16)
  float zf = z + __shfl_xor(z, 16, 32);    // zf at lane l == Z_{l&15}
  float bias = biasP[0];
  size_t outBase = (size_t)b * N_ * D_;
#pragma unroll
  for (int r = 0; r < 8; ++r) {
    int row  = rb + itile * 16 + r + 8 * hi;           // accumulator row layout
    float Zr = __shfl(zf, r + 8 * hi, 32);             // ds_bpermute broadcast
    float inv = 1.0f / Zr;
    float* o = out + outBase + (size_t)row * D_ + dq * 64 + ln;
    o[0]  = fmaxf(acc0[r] * inv + bias, 0.f);
    o[16] = fmaxf(acc1[r] * inv + bias, 0.f);
    o[32] = fmaxf(acc2[r] * inv + bias, 0.f);
    o[48] = fmaxf(acc3[r] * inv + bias, 0.f);
  }
}

// ---------------------------------------------------------------------------
// Launch. Workspace layout (bytes):
//   aArr 128K | bRaw 128K | Bp 128K | Bn 128K | cB 256 | Wt 128K | SFt 16M
// ---------------------------------------------------------------------------
extern "C" void kernel_launch(void* const* d_in, const int* in_sizes, int n_in,
                              void* d_out, int out_size, void* d_ws, size_t ws_size,
                              hipStream_t stream) {
  const float* seq  = (const float*)d_in[0];
  const float* W    = (const float*)d_in[1];
  const float* w1   = (const float*)d_in[2];
  const float* b1   = (const float*)d_in[3];
  const float* w2   = (const float*)d_in[4];
  const float* b2   = (const float*)d_in[5];
  const float* bias = (const float*)d_in[6];
  float* out = (float*)d_out;

  char* ws = (char*)d_ws;
  float*    aArr = (float*)(ws);
  float*    bRaw = (float*)(ws + 131072);
  float*    Bp   = (float*)(ws + 262144);
  float*    Bn   = (float*)(ws + 393216);
  float*    cB   = (float*)(ws + 524288);
  _Float16* Wt   = (_Float16*)(ws + 524544);
  _Float16* SFt  = (_Float16*)(ws + 655616);

  prep_w_kernel      <<<dim3(256),  dim3(256), 0, stream>>>(W, Wt);
  row_proj_kernel    <<<dim3(4096), dim3(256), 0, stream>>>(seq, w1, b1, w2, b2,
                                                            aArr, bRaw, Bp, Bn);
  batch_reduce_kernel<<<dim3(B_),   dim3(256), 0, stream>>>(aArr, bRaw, cB);
  gemm1_kernel       <<<dim3(4096), dim3(256), 0, stream>>>(seq, Wt, SFt);
  attn_kernel        <<<dim3(1024), dim3(256), 0, stream>>>(aArr, Bp, Bn, cB, SFt, bias, out);
}
